// PtrAttention_63548336112291
// MI455X (gfx1250) — compile-verified, matrix operand order checked by reading, not run on hardware
//
#include <hip/hip_runtime.h>
#include <hip/hip_bf16.h>
#include <cstddef>

typedef __attribute__((ext_vector_type(16))) _Float16 v16h;
typedef __attribute__((ext_vector_type(8)))  _Float16 v8h;
typedef __attribute__((ext_vector_type(8)))  float    v8f;

#define S_DIM 16384
#define H_DIM 1024
#define A_DIM 1024

#define BLK_M 256      // 8 waves x 2 M-tiles x 16 rows
#define BLK_N 128      // 8 N-tiles of 16 cols, B shared via LDS
#define KSTEP 32
#define MT    2        // M-tiles per wave (share B fragments -> halve LDS traffic/WMMA)
#define BS_STRIDE 40   // padded f16 stride (80B = 20 banks) -> conflict-free column reads

__device__ __forceinline__ v8h cvt8(const float4 a, const float4 b) {
    v8h r;
    r[0] = (_Float16)a.x; r[1] = (_Float16)a.y; r[2] = (_Float16)a.z; r[3] = (_Float16)a.w;
    r[4] = (_Float16)b.x; r[5] = (_Float16)b.y; r[6] = (_Float16)b.z; r[7] = (_Float16)b.w;
    return r;
}

__device__ __forceinline__ void loadW(const float* __restrict__ base, int kk, float4 o[4]) {
    const float4* q = (const float4*)(base + kk);
    o[0] = q[0]; o[1] = q[1]; o[2] = q[2]; o[3] = q[3];
}

// A fragment f32 source: lane<16 holds K {0..7,16..23}, lane>=16 holds {8..15,24..31}
__device__ __forceinline__ void loadA(const float* __restrict__ arow, int kk, int half,
                                      float4 o[4]) {
    const float4* p0 = (const float4*)(arow + kk + half * 8);
    const float4* p1 = (const float4*)(arow + kk + 16 + half * 8);
    o[0] = p0[0]; o[1] = p0[1]; o[2] = p1[0]; o[3] = p1[1];
}

__device__ __forceinline__ v16h cvtA(const float4 o[4]) {
    v8h lo = cvt8(o[0], o[1]);
    v8h hi = cvt8(o[2], o[3]);
    v16h r;
#pragma unroll
    for (int j = 0; j < 8; ++j) { r[j] = lo[j]; r[8 + j] = hi[j]; }
    return r;
}

// ---------------------------------------------------------------------------
// Kernel 1: d = dec_h @ W2^T   (one wave per output element)
// ---------------------------------------------------------------------------
__global__ __launch_bounds__(256)
void prep_d_kernel(const float* __restrict__ dec_h, const float* __restrict__ W2,
                   float* __restrict__ d_vec) {
    const int wave = threadIdx.x >> 5, lane = threadIdx.x & 31;
    const int a = blockIdx.x * 8 + wave;
    float s = 0.f;
    const float* wrow = W2 + (size_t)a * H_DIM;
    for (int h = lane; h < H_DIM; h += 32) s += dec_h[h] * wrow[h];
    s += __shfl_xor(s, 1);  s += __shfl_xor(s, 2);  s += __shfl_xor(s, 4);
    s += __shfl_xor(s, 8);  s += __shfl_xor(s, 16);
    if (lane == 0) d_vec[a] = s;
}

// ---------------------------------------------------------------------------
// Kernel 2: zero the u accumulator (workspace is poisoned, atomics need 0)
// ---------------------------------------------------------------------------
__global__ void zero_u_kernel(float* __restrict__ u_acc) {
    int i = blockIdx.x * blockDim.x + threadIdx.x;
    if (i < S_DIM) u_acc[i] = 0.f;
}

// ---------------------------------------------------------------------------
// Kernel 3 (main): e = enc @ W1^T via v_wmma_f32_16x16x32_f16, fused with
// u[s] += sum_a tanh(d[a] + e[s,a]) * V[a]  (partial over this block's cols).
// Double-buffered LDS, software-pipelined global loads, 2 M-tiles per wave.
// ---------------------------------------------------------------------------
__global__ __launch_bounds__(256)
void attn_gemm_tanh_dot(const float* __restrict__ enc,
                        const float* __restrict__ W1,
                        const float* __restrict__ d_vec,
                        const float* __restrict__ Vv,
                        float* __restrict__ u_acc) {
    __shared__ __align__(16) _Float16 Bs[2][BLK_N * BS_STRIDE];

    const int tid  = threadIdx.x;
    const int wave = tid >> 5;
    const int lane = tid & 31;
    const int lrow = lane & 15;
    const int half = lane >> 4;

    const int rowBase = blockIdx.x * BLK_M + wave * 16;  // M-tile 0; tile 1 at +128
    const int nBase   = blockIdx.y * BLK_N;

    const float* arow0 = enc + (size_t)(rowBase + lrow) * H_DIM;
    const float* arow1 = arow0 + (size_t)128 * H_DIM;

    v8f acc[MT][8];
#pragma unroll
    for (int mt = 0; mt < MT; ++mt)
#pragma unroll
        for (int nt = 0; nt < 8; ++nt)
#pragma unroll
            for (int r = 0; r < 8; ++r) acc[mt][nt][r] = 0.0f;

    // LDS staging assignment: thread t converts 16 contiguous f32 of W1 -> f16
    const int stg_n = tid >> 1;
    const int stg_k = (tid & 1) * 16;
    const float* stg_src = W1 + (size_t)(nBase + stg_n) * H_DIM + stg_k;

    // Pipeline prologue: loads for kk = 0 in flight
    float4 gw[4], ar0[4], ar1[4];
    loadW(stg_src, 0, gw);
    loadA(arow0, 0, half, ar0);
    loadA(arow1, 0, half, ar1);

    int buf = 0;
    for (int kk = 0; kk < H_DIM; kk += KSTEP, buf ^= 1) {
        // commit staged W1 tile to LDS buffer `buf`
        v8h* dst = (v8h*)&Bs[buf][stg_n * BS_STRIDE + stg_k];
        dst[0] = cvt8(gw[0], gw[1]);
        dst[1] = cvt8(gw[2], gw[3]);

        const int kn = kk + KSTEP;                 // uniform branch: EXEC unchanged
        if (kn < H_DIM) loadW(stg_src, kn, gw);    // next W1 tile in flight during WMMAs

        v16h af0 = cvtA(ar0);
        v16h af1 = cvtA(ar1);
        if (kn < H_DIM) { loadA(arow0, kn, half, ar0); loadA(arow1, kn, half, ar1); }

        __syncthreads();   // single barrier/iter is safe with double buffering

#pragma unroll
        for (int nt = 0; nt < 8; ++nt) {
            const _Float16* bp = &Bs[buf][(nt * 16 + lrow) * BS_STRIDE];
            v8h blo = *(const v8h*)(bp + half * 8);
            v8h bhi = *(const v8h*)(bp + 16 + half * 8);
            v16h bfrag;
#pragma unroll
            for (int j = 0; j < 8; ++j) { bfrag[j] = blo[j]; bfrag[8 + j] = bhi[j]; }
            acc[0][nt] = __builtin_amdgcn_wmma_f32_16x16x32_f16(
                false, af0, false, bfrag, (short)0, acc[0][nt], false, false);
            acc[1][nt] = __builtin_amdgcn_wmma_f32_16x16x32_f16(
                false, af1, false, bfrag, (short)0, acc[1][nt], false, false);
        }
    }

    // Epilogue: per-lane column contribution, reduce across the 16-lane half.
    // C/D layout: VGPR r, lane l -> row M = r + 8*(l>=16), col N = l&15.
#pragma unroll
    for (int mt = 0; mt < MT; ++mt) {
        float us[8];
#pragma unroll
        for (int r = 0; r < 8; ++r) us[r] = 0.0f;
#pragma unroll
        for (int nt = 0; nt < 8; ++nt) {
            const int col = nBase + nt * 16 + lrow;
            const float dv = d_vec[col];
            const float vv = Vv[col];
#pragma unroll
            for (int r = 0; r < 8; ++r) us[r] += tanhf(dv + acc[mt][nt][r]) * vv;
        }
#pragma unroll
        for (int r = 0; r < 8; ++r) {
            float v = us[r];
            v += __shfl_xor(v, 1);
            v += __shfl_xor(v, 2);
            v += __shfl_xor(v, 4);
            v += __shfl_xor(v, 8);
            if (lrow == 0)
                atomicAdd(&u_acc[rowBase + mt * 128 + r + half * 8], v);
        }
    }
}

// ---------------------------------------------------------------------------
// Kernel 4: apply mask, write u output, compute softmax max & denom,
// zero the context-vector output region.
// ---------------------------------------------------------------------------
__global__ __launch_bounds__(1024)
void softmax_stats_kernel(const float* __restrict__ u_acc, const int* __restrict__ mask,
                          float* __restrict__ u_out, float* __restrict__ out_vec,
                          float* __restrict__ red) {
    __shared__ float sred[32];
    const int tid = threadIdx.x;
    const int lane = tid & 31, wid = tid >> 5;
    const float NEG_INF = __uint_as_float(0xff800000u);

    out_vec[tid] = 0.f;  // init context output for kernel 5's atomics

    float mx = NEG_INF;
    for (int s = tid; s < S_DIM; s += 1024) {
        float u = u_acc[s];
        if (mask[s] != 0) u = NEG_INF;
        u_out[s] = u;
        mx = fmaxf(mx, u);
    }
    mx = fmaxf(mx, __shfl_xor(mx, 1));  mx = fmaxf(mx, __shfl_xor(mx, 2));
    mx = fmaxf(mx, __shfl_xor(mx, 4));  mx = fmaxf(mx, __shfl_xor(mx, 8));
    mx = fmaxf(mx, __shfl_xor(mx, 16));
    if (lane == 0) sred[wid] = mx;
    __syncthreads();
    if (wid == 0) {
        float v = sred[lane];
        v = fmaxf(v, __shfl_xor(v, 1));  v = fmaxf(v, __shfl_xor(v, 2));
        v = fmaxf(v, __shfl_xor(v, 4));  v = fmaxf(v, __shfl_xor(v, 8));
        v = fmaxf(v, __shfl_xor(v, 16));
        if (lane == 0) sred[0] = v;
    }
    __syncthreads();
    mx = sred[0];
    __syncthreads();

    float su = 0.f;
    for (int s = tid; s < S_DIM; s += 1024) su += expf(u_out[s] - mx);  // exp(-inf)=0
    su += __shfl_xor(su, 1);  su += __shfl_xor(su, 2);  su += __shfl_xor(su, 4);
    su += __shfl_xor(su, 8);  su += __shfl_xor(su, 16);
    if (lane == 0) sred[wid] = su;
    __syncthreads();
    if (wid == 0) {
        float v = sred[lane];
        v += __shfl_xor(v, 1);  v += __shfl_xor(v, 2);  v += __shfl_xor(v, 4);
        v += __shfl_xor(v, 8);  v += __shfl_xor(v, 16);
        if (lane == 0) { red[0] = mx; red[1] = v; }
    }
}

// ---------------------------------------------------------------------------
// Kernel 5: out[h] = sum_s softmax(u)[s] * enc[s,h]   (memory-bound pass)
// ---------------------------------------------------------------------------
__global__ __launch_bounds__(256)
void weighted_sum_kernel(const float* __restrict__ enc, const float* __restrict__ u_out,
                         const float* __restrict__ red, float* __restrict__ out_vec) {
    __shared__ float wa[128];
    const int tid = threadIdx.x;
    const float mx = red[0];
    const float invDen = 1.0f / red[1];
    const int sBase = blockIdx.x * 128;
    if (tid < 128) wa[tid] = expf(u_out[sBase + tid] - mx) * invDen;
    __syncthreads();
    float acc[4] = {0.f, 0.f, 0.f, 0.f};
    for (int s = 0; s < 128; ++s) {
        const float w = wa[s];
        if (w != 0.f) {  // uniform branch: skip masked rows entirely
            const float* rp = enc + (size_t)(sBase + s) * H_DIM;
#pragma unroll
            for (int j = 0; j < 4; ++j) acc[j] += w * rp[tid + 256 * j];
        }
    }
#pragma unroll
    for (int j = 0; j < 4; ++j) atomicAdd(&out_vec[tid + 256 * j], acc[j]);
}

// ---------------------------------------------------------------------------
extern "C" void kernel_launch(void* const* d_in, const int* in_sizes, int n_in,
                              void* d_out, int out_size, void* d_ws, size_t ws_size,
                              hipStream_t stream) {
    (void)in_sizes; (void)n_in; (void)out_size; (void)ws_size;

    const float* enc   = (const float*)d_in[0];  // [S,1,H] -> [S,H]
    const float* dec_h = (const float*)d_in[1];  // [1,H]
    const int*   mask  = (const int*)d_in[2];    // [S,1] bool->int
    const float* W2    = (const float*)d_in[3];  // [A,H]
    const float* W1    = (const float*)d_in[4];  // [A,H]
    const float* Vv    = (const float*)d_in[5];  // [1,A]

    float* out   = (float*)d_out;     // [0..H)       context vector
    float* u_out = out + H_DIM;       // [H..H+S)     masked logits output

    float* W     = (float*)d_ws;
    float* d_vec = W;                 // 1024 floats
    float* u_acc = W + 1024;          // 16384 floats
    float* red   = W + 1024 + S_DIM;  // 2 floats (max, denom)

    prep_d_kernel<<<A_DIM / 8, 256, 0, stream>>>(dec_h, W2, d_vec);
    zero_u_kernel<<<(S_DIM + 255) / 256, 256, 0, stream>>>(u_acc);

    dim3 grid(S_DIM / BLK_M, A_DIM / BLK_N);  // (64, 8)
    attn_gemm_tanh_dot<<<grid, 256, 0, stream>>>(enc, W1, d_vec, Vv, u_acc);

    softmax_stats_kernel<<<1, 1024, 0, stream>>>(u_acc, mask, u_out, out, red);
    weighted_sum_kernel<<<S_DIM / 128, 256, 0, stream>>>(enc, u_out, red, out);
}
